// LSTM_Encoder_26508538151168
// MI455X (gfx1250) — matrix operand order repaired and checked by reading, not compiled, not run on hardware
//
#include <hip/hip_runtime.h>
#include <math.h>

// ---------------- problem constants ----------------
#define SEQ    512
#define BATCH  64
#define EMB    1024
#define HID    1024
#define G4     4096            // 4*HID
#define ROWS   (SEQ*BATCH)     // 32768
#define NWG    64              // workgroups in persistent recurrent kernel
#define COLS_PER_WG (HID/NWG)  // 16 hidden units per WG

typedef __attribute__((ext_vector_type(16))) __bf16 v16bf;
typedef __attribute__((ext_vector_type(8)))  __bf16 v8bf;
typedef __attribute__((ext_vector_type(8)))  float  v8f;

// ---------------- workspace layout (bytes) ----------------
static constexpr size_t OFF_X    = 0;                                  // 64 MiB
static constexpr size_t OFF_WIH  = OFF_X   + (size_t)ROWS*EMB*2;       //  8 MiB
static constexpr size_t OFF_WHH  = OFF_WIH + (size_t)G4*EMB*2;         //  8 MiB
static constexpr size_t OFF_BIAS = OFF_WHH + (size_t)G4*HID*2;
static constexpr size_t OFF_H    = OFF_BIAS + (size_t)G4*4;            // 2x h state (bf16)
static constexpr size_t OFF_BAR  = OFF_H   + (size_t)2*BATCH*HID*2;
static constexpr size_t OFF_PRE  = OFF_BAR + 256;                      // 512 MiB

__device__ __forceinline__ float sigf(float x) {
    return 1.0f / (1.0f + __expf(-x));
}

// ---------------- fp32 -> bf16 cast ----------------
__global__ void cast_f32_bf16(const float* __restrict__ src,
                              __bf16* __restrict__ dst, int n) {
    int i = (blockIdx.x * 256 + threadIdx.x) * 4;
    if (i + 3 < n) {
        float4 v = *(const float4*)(src + i);
        dst[i + 0] = (__bf16)v.x;
        dst[i + 1] = (__bf16)v.y;
        dst[i + 2] = (__bf16)v.z;
        dst[i + 3] = (__bf16)v.w;
    }
}

// ---------------- bias fuse: bias = b_ih + b_hh ----------------
__global__ void bias_fuse(const float* __restrict__ b_ih,
                          const float* __restrict__ b_hh,
                          float* __restrict__ bias) {
    int i = blockIdx.x * 256 + threadIdx.x;
    if (i < G4) bias[i] = b_ih[i] + b_hh[i];
}

// ---------------- embedding gather + cast: one block per token ----------------
__global__ void embed_gather(const float* __restrict__ embedding,
                             const int* __restrict__ tokens,
                             __bf16* __restrict__ X) {
    int row = blockIdx.x;                 // 0..32767 = s*BATCH + b
    int tok = tokens[row];
    const float* src = embedding + (size_t)tok * EMB;
    __bf16* dst = X + (size_t)row * EMB;
    int e = threadIdx.x * 4;              // 256 thr * 4 = 1024
    float4 v = *(const float4*)(src + e);
    dst[e + 0] = (__bf16)v.x;
    dst[e + 1] = (__bf16)v.y;
    dst[e + 2] = (__bf16)v.z;
    dst[e + 3] = (__bf16)v.w;
}

// ---------------- A fragment: 16x32 bf16, row-major source ----------------
__device__ __forceinline__ v16bf load_a_frag(const __bf16* base, int ld,
                                             int m0, int k0, int lane) {
    int row  = lane & 15;
    int half = lane >> 4;
    const __bf16* p = base + (size_t)(m0 + row) * ld + k0;
    v8bf lo = *(const v8bf*)(p + half * 16);       // K = half*16 .. +7
    v8bf hi = *(const v8bf*)(p + half * 16 + 8);   // K = half*16+8 .. +7
    return __builtin_shufflevector(lo, hi, 0,1,2,3,4,5,6,7,8,9,10,11,12,13,14,15);
}

// ---------------- B fragment: 32x16 bf16, B[k,n]=W[n,k], W row-major [N,K] ----------------
__device__ __forceinline__ v16bf load_b_frag(const __bf16* W, int ld,
                                             int n0, int k0, int lane) {
    int row  = lane & 15;
    int half = lane >> 4;
    return *(const v16bf*)(W + (size_t)(n0 + row) * ld + k0 + half * 16);
}

// ---------------- pre = X @ W_ih^T + bias  (32768x4096) ----------------
// grid: 512 M-blocks * 32 N-blocks; block = 256 thr = 8 waves.
// Ping-pong (2-deep) fragment buffers, stride-64 K loop: each refill targets a
// dead buffer so the allocator needs no cross-buffer copies (no v_mov chatter).
__global__ void __launch_bounds__(256)
gemm_pre(const __bf16* __restrict__ X, const __bf16* __restrict__ Wih,
         const float* __restrict__ bias, float* __restrict__ pre) {
    int nb = blockIdx.x & 31;           // N block (128 cols)
    int mb = blockIdx.x >> 5;           // M block (64 rows)
    int lane = threadIdx.x & 31;
    int wave = threadIdx.x >> 5;
    int mt = wave & 3;
    int ng = wave >> 2;
    int m0 = mb * 64 + mt * 16;
    int nbase = nb * 128 + ng * 64;

    v8f acc[4] = {};
    v16bf aP, aQ, bP[4], bQ[4];
    aP = load_a_frag(X, EMB, m0, 0, lane);
    aQ = load_a_frag(X, EMB, m0, 32, lane);
#pragma unroll
    for (int nt = 0; nt < 4; ++nt) {
        bP[nt] = load_b_frag(Wih, EMB, nbase + nt * 16, 0, lane);
        bQ[nt] = load_b_frag(Wih, EMB, nbase + nt * 16, 32, lane);
    }

    for (int k0 = 0; k0 < EMB - 64; k0 += 64) {
        // consume slice k0 (buffer P), refill P with slice k0+64
#pragma unroll
        for (int nt = 0; nt < 4; ++nt)
            acc[nt] = __builtin_amdgcn_wmma_f32_16x16x32_bf16(
                false, aP, false, bP[nt], (short)0, acc[nt], false, false);
        aP = load_a_frag(X, EMB, m0, k0 + 64, lane);
#pragma unroll
        for (int nt = 0; nt < 4; ++nt)
            bP[nt] = load_b_frag(Wih, EMB, nbase + nt * 16, k0 + 64, lane);
        __builtin_prefetch(Wih + (size_t)(nbase + (lane & 15)) * EMB + k0 + 320, 0, 1);
        // consume slice k0+32 (buffer Q), refill Q with slice k0+96
#pragma unroll
        for (int nt = 0; nt < 4; ++nt)
            acc[nt] = __builtin_amdgcn_wmma_f32_16x16x32_bf16(
                false, aQ, false, bQ[nt], (short)0, acc[nt], false, false);
        aQ = load_a_frag(X, EMB, m0, k0 + 96, lane);
#pragma unroll
        for (int nt = 0; nt < 4; ++nt)
            bQ[nt] = load_b_frag(Wih, EMB, nbase + nt * 16, k0 + 96, lane);
    }
    // epilogue: slices EMB-64 (P) and EMB-32 (Q)
#pragma unroll
    for (int nt = 0; nt < 4; ++nt)
        acc[nt] = __builtin_amdgcn_wmma_f32_16x16x32_bf16(
            false, aP, false, bP[nt], (short)0, acc[nt], false, false);
#pragma unroll
    for (int nt = 0; nt < 4; ++nt)
        acc[nt] = __builtin_amdgcn_wmma_f32_16x16x32_bf16(
            false, aQ, false, bQ[nt], (short)0, acc[nt], false, false);

    int half = lane >> 4;
#pragma unroll
    for (int nt = 0; nt < 4; ++nt) {
        int n = nbase + nt * 16 + (lane & 15);
        float bs = bias[n];
#pragma unroll
        for (int r = 0; r < 8; ++r) {
            int m = m0 + half * 8 + r;
            pre[(size_t)m * G4 + n] = acc[nt][r] + bs;
        }
    }
}

// ---------------- init: zero h0 buffer and grid barrier ----------------
__global__ void init_state(__bf16* hbuf, unsigned* bar) {
    int i = blockIdx.x * 256 + threadIdx.x;
    if (i < BATCH * HID) hbuf[i] = (__bf16)0.0f;
    if (i == 0) *bar = 0u;
}

// ---------------- persistent recurrent kernel: 512 sequential steps ----------------
// 64 WGs * 256 thr (8 waves). WG owns hidden cols [wg*16, wg*16+16).
// h_{t-1} (128 KiB bf16) is staged into LDS each step via async-to-LDS DMA,
// then A-fragments come from LDS (ds_load) while B streams from L2-resident W_hh
// through ping-pong buffers.
__global__ void __launch_bounds__(256)
lstm_steps(const __bf16* __restrict__ Whh, const float* __restrict__ pre,
           const int* __restrict__ s_mask, __bf16* __restrict__ hbuf,
           unsigned* __restrict__ bar, float* __restrict__ out) {
    __shared__ __bf16 hlds[BATCH * HID];          // 128 KiB staged h_{t-1}
    __shared__ float gl[4][BATCH][COLS_PER_WG];   // staged gate values (16 KiB)
    __shared__ float cl[BATCH][COLS_PER_WG];      // persistent cell state (4 KiB)

    const int wg   = blockIdx.x;
    const int tid  = threadIdx.x;
    const int lane = tid & 31;
    const int wave = tid >> 5;
    const int g    = wave >> 1;
    const int mp   = (wave & 1) * 2;              // M-tile pair base (tiles of 16)
    const int col0 = wg * COLS_PER_WG;
    const int row  = lane & 15;
    const int half = lane >> 4;

    for (int e = tid; e < BATCH * COLS_PER_WG; e += 256)
        cl[e >> 4][e & 15] = 0.0f;
    __syncthreads();

    // This wave's B rows: W_hh rows (g*1024 + col0 + row), contiguous in K.
    const __bf16* Wrow = Whh + (size_t)(g * HID + col0) * HID;
    // LDS byte offset of this thread's staging slice (generic->LDS offset truncation)
    const unsigned lds_base = (unsigned)(uintptr_t)(&hlds[0]) + (unsigned)tid * 512u;

    for (int t = 0; t < SEQ; ++t) {
        const __bf16* hprev = hbuf + (size_t)(t & 1) * BATCH * HID;
        __bf16*       hnext = hbuf + (size_t)((t + 1) & 1) * BATCH * HID;

        // ---- async DMA: stage h_{t-1} global -> LDS (512 B per thread) ----
        {
            const char* gsrc = (const char*)hprev + (size_t)tid * 512u;
#pragma unroll
            for (int i = 0; i < 32; ++i) {
                unsigned loff = lds_base + (unsigned)(i * 16);
                unsigned long long ga = (unsigned long long)(uintptr_t)(gsrc + i * 16);
                asm volatile("global_load_async_to_lds_b128 %0, %1, off"
                             :: "v"(loff), "v"(ga) : "memory");
            }
        }
        asm volatile("s_wait_asynccnt 0x0" ::: "memory");
        __syncthreads();

        // ---- recurrent GEMM slice: A from LDS, B from L2, ping-pong on B ----
        v8f acc0 = {}, acc1 = {};
        v16bf bP = load_b_frag(Wrow, HID, 0, 0, lane);
        v16bf bQ = load_b_frag(Wrow, HID, 0, 32, lane);
        for (int k0 = 0; k0 < HID - 64; k0 += 64) {
            {   // consume slice k0 (P), refill P with k0+64
                v16bf a0 = load_a_frag(hlds, HID, (mp + 0) * 16, k0, lane);
                v16bf a1 = load_a_frag(hlds, HID, (mp + 1) * 16, k0, lane);
                acc0 = __builtin_amdgcn_wmma_f32_16x16x32_bf16(
                    false, a0, false, bP, (short)0, acc0, false, false);
                acc1 = __builtin_amdgcn_wmma_f32_16x16x32_bf16(
                    false, a1, false, bP, (short)0, acc1, false, false);
                bP = load_b_frag(Wrow, HID, 0, k0 + 64, lane);
            }
            {   // consume slice k0+32 (Q), refill Q with k0+96
                v16bf a0 = load_a_frag(hlds, HID, (mp + 0) * 16, k0 + 32, lane);
                v16bf a1 = load_a_frag(hlds, HID, (mp + 1) * 16, k0 + 32, lane);
                acc0 = __builtin_amdgcn_wmma_f32_16x16x32_bf16(
                    false, a0, false, bQ, (short)0, acc0, false, false);
                acc1 = __builtin_amdgcn_wmma_f32_16x16x32_bf16(
                    false, a1, false, bQ, (short)0, acc1, false, false);
                bQ = load_b_frag(Wrow, HID, 0, k0 + 96, lane);
            }
        }
        {   // epilogue: slices HID-64 (P) and HID-32 (Q)
            v16bf a0 = load_a_frag(hlds, HID, (mp + 0) * 16, HID - 64, lane);
            v16bf a1 = load_a_frag(hlds, HID, (mp + 1) * 16, HID - 64, lane);
            acc0 = __builtin_amdgcn_wmma_f32_16x16x32_bf16(
                false, a0, false, bP, (short)0, acc0, false, false);
            acc1 = __builtin_amdgcn_wmma_f32_16x16x32_bf16(
                false, a1, false, bP, (short)0, acc1, false, false);
            a0 = load_a_frag(hlds, HID, (mp + 0) * 16, HID - 32, lane);
            a1 = load_a_frag(hlds, HID, (mp + 1) * 16, HID - 32, lane);
            acc0 = __builtin_amdgcn_wmma_f32_16x16x32_bf16(
                false, a0, false, bQ, (short)0, acc0, false, false);
            acc1 = __builtin_amdgcn_wmma_f32_16x16x32_bf16(
                false, a1, false, bQ, (short)0, acc1, false, false);
        }

        // add precomputed input gates + bias, stage in LDS
        const float* preT = pre + (size_t)t * BATCH * G4 + g * HID + col0 + row;
#pragma unroll
        for (int r = 0; r < 8; ++r) {
            int m0 = (mp + 0) * 16 + half * 8 + r;
            int m1 = (mp + 1) * 16 + half * 8 + r;
            gl[g][m0][row] = acc0[r] + preT[(size_t)m0 * G4];
            gl[g][m1][row] = acc1[r] + preT[(size_t)m1 * G4];
        }
        __syncthreads();

        // elementwise LSTM cell update for this WG's 64x16 slice
        for (int e = tid; e < BATCH * COLS_PER_WG; e += 256) {
            int m = e >> 4, cc = e & 15;
            float iv = sigf(gl[0][m][cc]);
            float fv = sigf(gl[1][m][cc]);
            float gv = tanhf(gl[2][m][cc]);
            float ov = sigf(gl[3][m][cc]);
            float cold = cl[m][cc];
            float hold = (float)hlds[m * HID + col0 + cc];
            float cn = fv * cold + iv * gv;
            float hn = ov * tanhf(cn);
            bool keep = (s_mask[t * BATCH + m] == 1);   // mask==1 -> keep old state
            float cf = keep ? cold : cn;
            float hf = keep ? hold : hn;
            cl[m][cc] = cf;
            hnext[(size_t)m * HID + col0 + cc] = (__bf16)hf;
            if (t == SEQ - 1) out[(size_t)m * HID + col0 + cc] = hf;
        }
        __syncthreads();

        // grid-wide release/acquire barrier (monotonic counter, 64 WGs resident)
        if (tid == 0) {
            __threadfence();
            atomicAdd(bar, 1u);
            unsigned target = (unsigned)NWG * (unsigned)(t + 1);
            while (__hip_atomic_load(bar, __ATOMIC_ACQUIRE,
                                     __HIP_MEMORY_SCOPE_AGENT) < target) {
                __builtin_amdgcn_s_sleep(1);
            }
        }
        __syncthreads();
    }
}

// ---------------- host side ----------------
extern "C" void kernel_launch(void* const* d_in, const int* in_sizes, int n_in,
                              void* d_out, int out_size, void* d_ws, size_t ws_size,
                              hipStream_t stream) {
    const float* embedding = (const float*)d_in[0];   // [32000,1024]
    const float* W_ih      = (const float*)d_in[1];   // [4096,1024]
    const float* W_hh      = (const float*)d_in[2];   // [4096,1024]
    const float* b_ih      = (const float*)d_in[3];   // [4096]
    const float* b_hh      = (const float*)d_in[4];   // [4096]
    const int*   inputs    = (const int*)d_in[5];     // [512,64]
    const int*   s_mask    = (const int*)d_in[6];     // [512,64]
    float* out = (float*)d_out;                       // [64,1024]

    char* ws = (char*)d_ws;
    __bf16*   X    = (__bf16*)(ws + OFF_X);
    __bf16*   Wihb = (__bf16*)(ws + OFF_WIH);
    __bf16*   Whhb = (__bf16*)(ws + OFF_WHH);
    float*    bias = (float*) (ws + OFF_BIAS);
    __bf16*   hbuf = (__bf16*)(ws + OFF_H);
    unsigned* bar  = (unsigned*)(ws + OFF_BAR);
    float*    pre  = (float*) (ws + OFF_PRE);

    // weights -> bf16 (L2-resident working set for both GEMMs)
    cast_f32_bf16<<<(G4 * EMB) / 1024, 256, 0, stream>>>(W_ih, Wihb, G4 * EMB);
    cast_f32_bf16<<<(G4 * HID) / 1024, 256, 0, stream>>>(W_hh, Whhb, G4 * HID);
    bias_fuse<<<(G4 + 255) / 256, 256, 0, stream>>>(b_ih, b_hh, bias);

    // embedding gather + cast
    embed_gather<<<ROWS, 256, 0, stream>>>(embedding, inputs, X);

    // bulk parallel GEMM: pre = X @ W_ih^T + (b_ih + b_hh)
    gemm_pre<<<(ROWS / 64) * (G4 / 128), 256, 0, stream>>>(X, Wihb, bias, pre);

    // state init + persistent sequential scan
    init_state<<<(BATCH * HID + 255) / 256, 256, 0, stream>>>(hbuf, bar);
    lstm_steps<<<NWG, 256, 0, stream>>>(Whhb, pre, s_mask, hbuf, bar, out);
}
